// RecurrentEncoder_13975823581757
// MI455X (gfx1250) — matrix operand order repaired and checked by reading, not compiled, not run on hardware
//
#include <hip/hip_runtime.h>
#include <hip/hip_bf16.h>
#include <hip/hip_fp16.h>

// ---------------------------------------------------------------------------
// Bidirectional 2-layer mLSTM encoder for MI455X (gfx1250, wave32, WMMA).
// T=256, B=64, E=H=512, L=2.
// ---------------------------------------------------------------------------

typedef _Float16 h16_t;
typedef __attribute__((ext_vector_type(16))) _Float16 v16h;
typedef __attribute__((ext_vector_type(8)))  _Float16 v8h;
typedef __attribute__((ext_vector_type(8)))  float    v8f;

#define TT   256
#define BB   64
#define EE   512
#define HH   512
#define NTOK (TT * BB)      // 16384 rows
#define G4   (4 * HH)       // 2048 gate columns
#define N1   (HH + G4)      // 2560 combined activation cols [xm | xg]

// ---- WMMA fragment helpers (16-bit, K=32, contiguous k = 16*hi + i used for both A and B) ----

// A-matrix 16x32: lane r=lane&15 -> row M=r; hi=lane>>4 selects K-groups
// element i -> K = k0 + hi*16 + i  (contiguous 32B per lane)
__device__ __forceinline__ v16h load_afrag(const h16_t* A, int lda, int r, int hi, int k0) {
    const h16_t* p = A + (size_t)r * lda + k0 + hi * 16;
    return *(const v16h*)(p);
}

// B-matrix 32x16 from N-major storage BT[n][k]: lane n=lane&15 -> col N=n;
// element i -> K = k0 + hi*16 + i  (contiguous 32B per lane)
__device__ __forceinline__ v16h load_bfrag(const h16_t* BT, int ldk, int n, int hi, int k0) {
    const h16_t* p = BT + (size_t)n * ldk + k0 + hi * 16;
    return *(const v16h*)(p);
}

__device__ __forceinline__ v8f wmma32(v16h a, v16h b, v8f c) {
    return __builtin_amdgcn_wmma_f32_16x16x32_f16(false, a, false, b, (short)0, c, false, false);
}

__device__ __forceinline__ v8f vzero8() {
    v8f z = {0.f,0.f,0.f,0.f,0.f,0.f,0.f,0.f};
    return z;
}

__device__ __forceinline__ float sigmoidf_(float x) { return 1.f / (1.f + expf(-x)); }

// ---------------------------------------------------------------------------
// Kernel 1: embedding gather + f32->f16 convert.  grid = NTOK blocks x 128.
// ---------------------------------------------------------------------------
__global__ __launch_bounds__(128) void embed_k(const int* __restrict__ tok,
                                               const float* __restrict__ tab,
                                               h16_t* __restrict__ out) {
    const int row = blockIdx.x;
    const int t = tok[row];
    const float4* src = (const float4*)(tab + (size_t)t * EE);
    h16_t* dst = out + (size_t)row * EE;
    const int e = threadIdx.x;             // 0..127, 4 elements each
    float4 v = src[e];
    dst[e*4+0] = (h16_t)v.x;  dst[e*4+1] = (h16_t)v.y;
    dst[e*4+2] = (h16_t)v.z;  dst[e*4+3] = (h16_t)v.w;
}

// ---------------------------------------------------------------------------
// Weight transpose+convert kernels (per dir/layer).  N-major f16 output.
// wcvt1 also builds the full-length bias (zeros over the xm half) so the
// GEMM epilogue is completely branch-free.
// ---------------------------------------------------------------------------
__global__ __launch_bounds__(128) void wcvt1_k(const float* __restrict__ Wmx,   // (E,H)
                                               const float* __restrict__ Wx,    // (E,4H)
                                               const float* __restrict__ bv,    // (4H)
                                               h16_t* __restrict__ WT,          // [N1][E]
                                               float* __restrict__ bfull) {     // [N1]
    const int n = blockIdx.x;                       // 0..2559
    if (threadIdx.x == 0) bfull[n] = (n < HH) ? 0.f : bv[n - HH];
    for (int k = threadIdx.x; k < EE; k += 128) {
        float w = (n < HH) ? Wmx[(size_t)k * HH + n]
                           : Wx[(size_t)k * G4 + (n - HH)];
        WT[(size_t)n * EE + k] = (h16_t)w;
    }
}

__global__ __launch_bounds__(128) void wcvt2_k(const float* __restrict__ Wmh,   // (H,H)
                                               const float* __restrict__ Wm,    // (H,4H)
                                               h16_t* __restrict__ WmhT,        // [H][H]
                                               h16_t* __restrict__ WmT) {       // [4H][H]
    const int n = blockIdx.x;                       // 0..2559
    for (int k = threadIdx.x; k < HH; k += 128) {
        if (n < HH) WmhT[(size_t)n * HH + k] = (h16_t)Wmh[(size_t)k * HH + n];
        else        WmT[(size_t)(n - HH) * HH + k] = (h16_t)Wm[(size_t)k * G4 + (n - HH)];
    }
}

// ---------------------------------------------------------------------------
// Kernel 2: precompute GEMM  C[NTOK x 2560] = X(f16) @ [Wmx | Wx] + bfull
// Each wave owns a 32x32 macro-tile (2 A-frags x 2 B-frags -> 4 WMMAs/k-step)
// for 2x register reuse vs one tile per wave.  Uniform, branch-free epilogue.
// grid = (NTOK/32, 2560/256), block = 256 (8 waves).
// ---------------------------------------------------------------------------
__global__ __launch_bounds__(256) void gemm_xw(const h16_t* __restrict__ X,
                                               const h16_t* __restrict__ WT,
                                               const float* __restrict__ bfull,
                                               h16_t* __restrict__ C) {
    const int wave = threadIdx.x >> 5;
    const int lane = threadIdx.x & 31;
    const int nl = lane & 15;
    const int hi = lane >> 4;
    const int mt2 = blockIdx.x;                      // 0..511  (32 rows)
    const int nt2 = blockIdx.y * 8 + wave;           // 0..79   (32 cols)
    const h16_t* Ab = X  + (size_t)(mt2 * 32) * EE;
    const h16_t* Bb = WT + (size_t)(nt2 * 32) * EE;

    v8f d00 = vzero8(), d01 = vzero8(), d10 = vzero8(), d11 = vzero8();
    #pragma unroll 4
    for (int kk = 0; kk < 16; ++kk) {
        v16h a0 = load_afrag(Ab,           EE, nl, hi, kk * 32);
        v16h a1 = load_afrag(Ab + 16 * EE, EE, nl, hi, kk * 32);
        v16h b0 = load_bfrag(Bb,           EE, nl, hi, kk * 32);
        v16h b1 = load_bfrag(Bb + 16 * EE, EE, nl, hi, kk * 32);
        d00 = wmma32(a0, b0, d00);
        d01 = wmma32(a0, b1, d01);
        d10 = wmma32(a1, b0, d10);
        d11 = wmma32(a1, b1, d11);
    }

    const int c0 = nt2 * 32 + nl;        // cols of n-tile 0
    const int c1 = c0 + 16;              // cols of n-tile 1
    const float bb0 = bfull[c0];
    const float bb1 = bfull[c1];
    #pragma unroll
    for (int v = 0; v < 8; ++v) {
        const size_t r0 = (size_t)(mt2 * 32 + hi * 8 + v) * N1;       // m-tile 0
        const size_t r1 = (size_t)(mt2 * 32 + 16 + hi * 8 + v) * N1;  // m-tile 1
        C[r0 + c0] = (h16_t)(d00[v] + bb0);
        C[r0 + c1] = (h16_t)(d01[v] + bb1);
        C[r1 + c0] = (h16_t)(d10[v] + bb0);
        C[r1 + c1] = (h16_t)(d11[v] + bb1);
    }
}

// ---------------------------------------------------------------------------
// Kernel 3: persistent recurrent scan. ONE workgroup (512 threads = 16 waves)
// owns the whole T=256 recurrence for one (direction, layer).
// LDS: h (64x512 f16) | m (64x512 f16) | c (64x512 f32) | msk (64 f32).
// Per step:  phase A: m = xm_t * (h @ Wmh)   -> LDS
//            phase B: gates = xg_t + m @ Wm  -> fused LSTM cell update
// Activations come from the combined buffer: xact[row][0:512]=xm,
// xact[row][512:2560]=xg (bias pre-added).
// ---------------------------------------------------------------------------
__global__ __launch_bounds__(512) void scan_k(const h16_t* __restrict__ xact,
                                              const h16_t* __restrict__ WmhT,
                                              const h16_t* __restrict__ WmT,
                                              const int*   __restrict__ tokens,
                                              int rev, int isLayer0,
                                              h16_t* __restrict__ hseq,   // layer0 out (f16)
                                              float* __restrict__ yout,   // layer1 out (f32, +dir*512)
                                              float* __restrict__ hfin,
                                              float* __restrict__ cfin) {
    extern __shared__ char smem[];
    h16_t* hS   = (h16_t*)(smem);                                 //  64*512 f16
    h16_t* mS   = (h16_t*)(smem + 64*512*2);                      //  64*512 f16
    float* cS   = (float*)(smem + 2*64*512*2);                    //  64*512 f32
    float* mskS = (float*)(smem + 2*64*512*2 + 64*512*4);         //  64 f32

    const int wave = threadIdx.x >> 5;
    const int lane = threadIdx.x & 31;
    const int nl = lane & 15;
    const int hi = lane >> 4;

    for (int i = threadIdx.x; i < 64*512; i += 512) { hS[i] = (h16_t)0.f; cS[i] = 0.f; }
    __syncthreads();

    for (int s = 0; s < TT; ++s) {
        const int t = rev ? (TT - 1 - s) : s;
        const int row0 = t * BB;

        if (threadIdx.x < BB)
            mskS[threadIdx.x] = (tokens[row0 + threadIdx.x] != 0) ? 1.f : 0.f;

        // pull this step's activations (xm+xg, 64 rows x 5120B) toward the WGP
        {
            const int r   = threadIdx.x >> 3;        // 0..63
            const int seg = threadIdx.x & 7;         // 0..7 x 640B
            const char* pf = (const char*)(xact + (size_t)(row0 + r) * N1) + seg * 640;
            __builtin_prefetch(pf +   0, 0, 1);
            __builtin_prefetch(pf + 128, 0, 1);
            __builtin_prefetch(pf + 256, 0, 1);
            __builtin_prefetch(pf + 384, 0, 1);
            __builtin_prefetch(pf + 512, 0, 1);
        }

        // -------- phase A: m = xm_t * (h @ Wmh) --------
        for (int jj = 0; jj < 2; ++jj) {
            const int nt = wave + jj * 16;                    // 0..31 output col tiles
            const h16_t* Bb = WmhT + (size_t)(nt * 16) * HH;
            for (int mp = 0; mp < 2; ++mp) {
                v8f d0 = vzero8(), d1 = vzero8();
                #pragma unroll 2
                for (int kk = 0; kk < 16; ++kk) {
                    v16h bf = load_bfrag(Bb, HH, nl, hi, kk * 32);
                    v16h a0 = load_afrag(hS + (size_t)(mp*32)      * HH, HH, nl, hi, kk * 32);
                    v16h a1 = load_afrag(hS + (size_t)(mp*32 + 16) * HH, HH, nl, hi, kk * 32);
                    d0 = wmma32(a0, bf, d0);
                    d1 = wmma32(a1, bf, d1);
                }
                const int col = nt * 16 + nl;
                #pragma unroll
                for (int v = 0; v < 8; ++v) {
                    const int b0 = mp*32 + hi*8 + v;
                    const int b1 = b0 + 16;
                    const float x0 = (float)xact[(size_t)(row0 + b0) * N1 + col];
                    const float x1 = (float)xact[(size_t)(row0 + b1) * N1 + col];
                    mS[(size_t)b0 * HH + col] = (h16_t)(d0[v] * x0);
                    mS[(size_t)b1 * HH + col] = (h16_t)(d1[v] * x1);
                }
            }
        }
        __syncthreads();

        // -------- phase B: gates = xg_t + m @ Wm, fused cell update --------
        for (int jj = 0; jj < 2; ++jj) {
            const int j = wave + jj * 16;                     // h-column tile 0..31
            for (int mp = 0; mp < 2; ++mp) {
                v8f D[4][2];
                #pragma unroll
                for (int g = 0; g < 4; ++g) { D[g][0] = vzero8(); D[g][1] = vzero8(); }

                #pragma unroll 2
                for (int kk = 0; kk < 16; ++kk) {
                    v16h a0 = load_afrag(mS + (size_t)(mp*32)      * HH, HH, nl, hi, kk * 32);
                    v16h a1 = load_afrag(mS + (size_t)(mp*32 + 16) * HH, HH, nl, hi, kk * 32);
                    #pragma unroll
                    for (int g = 0; g < 4; ++g) {
                        const h16_t* Bb = WmT + (size_t)(g * HH + j * 16) * HH;
                        v16h bf = load_bfrag(Bb, HH, nl, hi, kk * 32);
                        D[g][0] = wmma32(a0, bf, D[g][0]);
                        D[g][1] = wmma32(a1, bf, D[g][1]);
                    }
                }

                const int h = j * 16 + nl;                    // each lane: unique h col
                #pragma unroll
                for (int q = 0; q < 2; ++q) {
                    #pragma unroll
                    for (int v = 0; v < 8; ++v) {
                        const int bi = mp*32 + q*16 + hi*8 + v;
                        const size_t rg = (size_t)(row0 + bi) * N1 + HH;  // xg base
                        const float gi = D[0][q][v] + (float)xact[rg + 0*HH + h];
                        const float gf = D[1][q][v] + (float)xact[rg + 1*HH + h];
                        const float go = D[2][q][v] + (float)xact[rg + 2*HH + h];
                        const float gg = D[3][q][v] + (float)xact[rg + 3*HH + h];
                        const float ig = sigmoidf_(gi);
                        const float fg = sigmoidf_(gf);
                        const float og = sigmoidf_(go);
                        const float tg = tanhf(gg);
                        const float co = cS[(size_t)bi * HH + h];
                        const float cn = fg * co + ig * tg;
                        const float hn = og * tanhf(cn);
                        const float mk = mskS[bi];
                        const float cst = mk * cn + (1.f - mk) * co;
                        const float ho  = (float)hS[(size_t)bi * HH + h];
                        const float hst = mk * hn + (1.f - mk) * ho;
                        cS[(size_t)bi * HH + h] = cst;
                        hS[(size_t)bi * HH + h] = (h16_t)hst;
                        if (isLayer0) hseq[(size_t)(row0 + bi) * HH + h] = (h16_t)hst;
                        else          yout[(size_t)(row0 + bi) * (2*HH) + h] = hst;
                        if (s == TT - 1) {
                            hfin[(size_t)bi * (2*HH) + h] = hst;
                            cfin[(size_t)bi * (2*HH) + h] = cst;
                        }
                    }
                }
            }
        }
        __syncthreads();
    }
}

// ---------------------------------------------------------------------------
// Host-side orchestration (all launches on `stream`; graph-capture safe).
// ---------------------------------------------------------------------------
extern "C" void kernel_launch(void* const* d_in, const int* in_sizes, int n_in,
                              void* d_out, int out_size, void* d_ws, size_t ws_size,
                              hipStream_t stream) {
    (void)in_sizes; (void)n_in; (void)out_size; (void)ws_size;

    const int*   tokens = (const int*)d_in[0];
    const float* tab    = (const float*)d_in[1];
    const float* Wmx[2] = { (const float*)d_in[2], (const float*)d_in[7]  };
    const float* Wmh[2] = { (const float*)d_in[3], (const float*)d_in[8]  };
    const float* Wx[2]  = { (const float*)d_in[4], (const float*)d_in[9]  };
    const float* Wm[2]  = { (const float*)d_in[5], (const float*)d_in[10] };
    const float* bv[2]  = { (const float*)d_in[6], (const float*)d_in[11] };

    float* out = (float*)d_out;
    float* ht  = out;                              // (L,B,2H) = 131072
    float* ct  = out + (size_t)2 * BB * 2 * HH;    // (L,B,2H)
    float* ys  = ct  + (size_t)2 * BB * 2 * HH;    // (T,B,2H)

    // workspace carving (all 256B aligned)
    char* ws = (char*)d_ws;
    size_t off = 0;
    h16_t* emb16  = (h16_t*)(ws + off); off += (size_t)NTOK * EE * 2;  // 16 MB
    h16_t* hseq16 = (h16_t*)(ws + off); off += (size_t)NTOK * HH * 2;  // 16 MB
    h16_t* xact16 = (h16_t*)(ws + off); off += (size_t)NTOK * N1 * 2;  // 80 MB
    h16_t* WT1    = (h16_t*)(ws + off); off += (size_t)N1 * EE * 2;    // 2.6 MB
    h16_t* WmhT   = (h16_t*)(ws + off); off += (size_t)HH * HH * 2;    // 0.5 MB
    h16_t* WmT    = (h16_t*)(ws + off); off += (size_t)G4 * HH * 2;    // 2 MB
    float* bfull  = (float*)(ws + off); off += (size_t)N1 * 4;         // 10 KB

    const size_t shbytes = (size_t)2 * 64 * 512 * 2 + (size_t)64 * 512 * 4 + 64 * 4; // 262400
    (void)hipFuncSetAttribute((const void*)scan_k,
                              hipFuncAttributeMaxDynamicSharedMemorySize,
                              (int)shbytes);

    embed_k<<<NTOK, 128, 0, stream>>>(tokens, tab, emb16);

    for (int d = 0; d < 2; ++d) {
        for (int l = 0; l < 2; ++l) {
            const float* pWmx = Wmx[d] + (size_t)l * EE * HH;
            const float* pWmh = Wmh[d] + (size_t)l * HH * HH;
            const float* pWx  = Wx[d]  + (size_t)l * EE * G4;
            const float* pWm  = Wm[d]  + (size_t)l * HH * G4;
            const float* pb   = bv[d]  + (size_t)l * G4;

            wcvt1_k<<<N1, 128, 0, stream>>>(pWmx, pWx, pb, WT1, bfull);
            wcvt2_k<<<N1, 128, 0, stream>>>(pWmh, pWm, WmhT, WmT);

            const h16_t* Xin = (l == 0) ? emb16 : hseq16;
            gemm_xw<<<dim3(NTOK / 32, 10), 256, 0, stream>>>(Xin, WT1, bfull, xact16);

            scan_k<<<dim3(1), dim3(512), shbytes, stream>>>(
                xact16, WmhT, WmT, tokens,
                /*rev=*/d, /*isLayer0=*/(l == 0) ? 1 : 0,
                hseq16,
                ys + (size_t)d * HH,
                ht + (size_t)l * BB * 2 * HH + (size_t)d * HH,
                ct + (size_t)l * BB * 2 * HH + (size_t)d * HH);
        }
    }
}
